// facevoice_memorynet_75428215652524
// MI455X (gfx1250) — matrix operand order repaired and checked by reading, not compiled, not run on hardware
//
#include <hip/hip_runtime.h>
#include <hip/hip_bf16.h>
#include <math.h>

typedef __bf16 bf16;
typedef __attribute__((ext_vector_type(16))) __bf16 v16bf;
typedef __attribute__((ext_vector_type(8)))  __bf16 v8bf;
typedef __attribute__((ext_vector_type(4)))  __bf16 v4bf;
typedef __attribute__((ext_vector_type(8)))  float  v8f;

#define C_DIM 256
#define COS_EPS 1e-8f

// ---------------------------------------------------------------------------
// fp32 -> bf16 cast + per-row L2 norm.  One block (64 threads) per 256-col row.
// ---------------------------------------------------------------------------
__global__ void cast_norm_kernel(const float* __restrict__ in, bf16* __restrict__ outb,
                                 float* __restrict__ norm, int rows) {
  int r = blockIdx.x;
  if (r >= rows) return;
  int t = threadIdx.x; // 64 threads, 4 elems each
  float4 v = ((const float4*)(in + (size_t)r * C_DIM))[t];
  float s = v.x * v.x + v.y * v.y + v.z * v.z + v.w * v.w;
  v4bf b;
  b[0] = (bf16)v.x; b[1] = (bf16)v.y; b[2] = (bf16)v.z; b[3] = (bf16)v.w;
  ((v4bf*)(outb + (size_t)r * C_DIM))[t] = b;
  __shared__ float red[64];
  red[t] = s; __syncthreads();
  for (int w = 32; w > 0; w >>= 1) { if (t < w) red[t] += red[t + w]; __syncthreads(); }
  if (t == 0) norm[r] = sqrtf(red[0]);
}

// ---------------------------------------------------------------------------
// Transposed split-bf16 copy of V:  hi[c*S+s] = bf16(V[s,c]),
// lo[c*S+s] = bf16(V[s,c] - float(hi)).   Coalesced writes, L2-served reads.
// ---------------------------------------------------------------------------
__global__ void transpose_split_kernel(const float* __restrict__ in,
                                       bf16* __restrict__ hi, bf16* __restrict__ lo, int S) {
  size_t o = (size_t)blockIdx.x * blockDim.x + threadIdx.x;
  size_t total = (size_t)S * C_DIM;
  if (o >= total) return;
  int c = (int)(o / S);
  int s = (int)(o % S);
  float v = in[(size_t)s * C_DIM + c];
  bf16 h = (bf16)v;
  hi[o] = h;
  lo[o] = (bf16)(v - (float)h);
}

// ---------------------------------------------------------------------------
// Cosine-logits GEMM: out[i,j] = (A_i . B_j) / max(|A_i||B_j|, eps)
// A: [N,C] bf16 row-major, B: [S,C] bf16 row-major (B used transposed).
// One wave per 16(M) x 64(N) tile, K = 256 fully unrolled -> 32 WMMAs.
// ---------------------------------------------------------------------------
__global__ void logits_wmma_kernel(const bf16* __restrict__ A, const bf16* __restrict__ B,
                                   const float* __restrict__ nA, const float* __restrict__ nB,
                                   float* __restrict__ out, int N, int S) {
  int wid  = (int)(((size_t)blockIdx.x * blockDim.x + threadIdx.x) >> 5);
  int lane = threadIdx.x & 31;
  int nb64 = S >> 6;
  int nWaves = (N >> 4) * nb64;
  if (wid >= nWaves) return;
  int bm = wid / nb64, bn = wid % nb64;
  int row0 = bm << 4, col0 = bn << 6;
  int m16 = lane & 15, kh = lane >> 4;

  const bf16* Ap = A + (size_t)(row0 + m16) * C_DIM;

  v8f acc[4];
#pragma unroll
  for (int t = 0; t < 4; t++)
#pragma unroll
    for (int i = 0; i < 8; i++) acc[t][i] = 0.0f;

#pragma unroll
  for (int k0 = 0; k0 < C_DIM; k0 += 32) {
    // A fragment: lane (m16, kh): elems[0..15]=A[m,k0+kh*16+..] (contiguous)
    v8bf alo = *(const v8bf*)(Ap + k0 + kh * 16);
    v8bf ahi = *(const v8bf*)(Ap + k0 + kh * 16 + 8);
    v16bf a;
#pragma unroll
    for (int i = 0; i < 8; i++) { a[i] = alo[i]; a[i + 8] = ahi[i]; }
#pragma unroll
    for (int t = 0; t < 4; t++) {
      // B fragment: lane column n = col0+t*16+m16, K-half kh: 16 contiguous bf16
      const bf16* Bp = B + (size_t)(col0 + t * 16 + m16) * C_DIM + k0 + kh * 16;
      v16bf b = *(const v16bf*)Bp;
      acc[t] = __builtin_amdgcn_wmma_f32_16x16x32_bf16(false, a, false, b, (short)0,
                                                       acc[t], false, false);
    }
  }

  float nam[8];
#pragma unroll
  for (int v = 0; v < 8; v++) nam[v] = nA[row0 + kh * 8 + v];
#pragma unroll
  for (int t = 0; t < 4; t++) {
    int n = col0 + t * 16 + m16;
    float nb = nB[n];
#pragma unroll
    for (int v = 0; v < 8; v++) {
      float denom = fmaxf(nam[v] * nb, COS_EPS);
      out[(size_t)(row0 + kh * 8 + v) * S + n] = acc[t][v] / denom;
    }
  }
}

// ---------------------------------------------------------------------------
// Per-row log-softmax IN PLACE: x <- x - (max + log(sum exp(x-max))).
// One 256-thread block per row; row stays cache-hot across the three passes.
// ---------------------------------------------------------------------------
__global__ void row_logsoftmax_kernel(float* __restrict__ raw, int S) {
  int r = blockIdx.x, t = threadIdx.x;
  float* p = raw + (size_t)r * S;
  float m = -3.4e38f;
  for (int i = t; i < S; i += 256) m = fmaxf(m, p[i]);
  __shared__ float red[256];
  red[t] = m; __syncthreads();
  for (int w = 128; w > 0; w >>= 1) { if (t < w) red[t] = fmaxf(red[t], red[t + w]); __syncthreads(); }
  float M = red[0];
  __syncthreads();
  float s = 0.0f;
  for (int i = t; i < S; i += 256) s += __expf(p[i] - M);
  red[t] = s; __syncthreads();
  for (int w = 128; w > 0; w >>= 1) { if (t < w) red[t] += red[t + w]; __syncthreads(); }
  float offv = M + __logf(red[0]);
  __syncthreads();
  for (int i = t; i < S; i += 256) p[i] -= offv;
}

// ---------------------------------------------------------------------------
// Recall = softmax @ V with bf16x3 compensated products.
// lp (log-softmax) is read-only here; p = exp(lp) split into hi+lo bf16,
// V^T pre-split into hi+lo bf16.  acc += ahi*bhi + ahi*blo + alo*bhi (fp32).
// One 128-thread block (4 waves) per 16-row strip; wave w owns columns
// [64w, 64w+64).  The 4 waves share the strip via the WGP-local cache.
// ---------------------------------------------------------------------------
__global__ void recall_wmma_kernel(const float* __restrict__ lp0, float* __restrict__ rec0,
                                   const float* __restrict__ lp1, float* __restrict__ rec1,
                                   const bf16* __restrict__ VThi, const bf16* __restrict__ VTlo,
                                   int N, int S) {
  int rbPerPair = N >> 4;
  int b = blockIdx.x;
  if (b >= 2 * rbPerPair) return;
  int pair = b & 1;
  int rb   = b >> 1;
  const float* lp = pair ? lp1 : lp0;
  float*       rec = pair ? rec1 : rec0;

  int lane = threadIdx.x & 31;
  int wave = threadIdx.x >> 5;           // 0..3
  int colbase = wave << 6;               // 64 columns per wave
  int row0 = rb << 4;
  int m16 = lane & 15, kh = lane >> 4;
  const float* Rp = lp + (size_t)(row0 + m16) * S;

  v8f acc[4];
#pragma unroll
  for (int t = 0; t < 4; t++)
#pragma unroll
    for (int i = 0; i < 8; i++) acc[t][i] = 0.0f;

  for (int k0 = 0; k0 < S; k0 += 32) {
    if (k0 + 160 <= S) __builtin_prefetch(Rp + k0 + 128, 0, 0);
    const float4* plo4 = (const float4*)(Rp + k0 + kh * 16);
    const float4* phi4 = (const float4*)(Rp + k0 + kh * 16 + 8);
    float4 xs[4];
    xs[0] = plo4[0]; xs[1] = plo4[1]; xs[2] = phi4[0]; xs[3] = phi4[1];
    v16bf ahi, alo;
#pragma unroll
    for (int q = 0; q < 4; q++) {
      float p0 = __expf(xs[q].x), p1 = __expf(xs[q].y);
      float p2 = __expf(xs[q].z), p3 = __expf(xs[q].w);
      bf16 h0 = (bf16)p0, h1 = (bf16)p1, h2 = (bf16)p2, h3 = (bf16)p3;
      ahi[4 * q + 0] = h0; ahi[4 * q + 1] = h1; ahi[4 * q + 2] = h2; ahi[4 * q + 3] = h3;
      alo[4 * q + 0] = (bf16)(p0 - (float)h0);
      alo[4 * q + 1] = (bf16)(p1 - (float)h1);
      alo[4 * q + 2] = (bf16)(p2 - (float)h2);
      alo[4 * q + 3] = (bf16)(p3 - (float)h3);
    }
#pragma unroll
    for (int t = 0; t < 4; t++) {
      size_t boff = (size_t)(colbase + t * 16 + m16) * S + k0 + kh * 16;
      v16bf bhi = *(const v16bf*)(VThi + boff);
      v16bf blo = *(const v16bf*)(VTlo + boff);
      acc[t] = __builtin_amdgcn_wmma_f32_16x16x32_bf16(false, ahi, false, bhi, (short)0,
                                                       acc[t], false, false);
      acc[t] = __builtin_amdgcn_wmma_f32_16x16x32_bf16(false, ahi, false, blo, (short)0,
                                                       acc[t], false, false);
      acc[t] = __builtin_amdgcn_wmma_f32_16x16x32_bf16(false, alo, false, bhi, (short)0,
                                                       acc[t], false, false);
    }
  }

#pragma unroll
  for (int t = 0; t < 4; t++) {
    int n = colbase + t * 16 + m16;
#pragma unroll
    for (int v = 0; v < 8; v++)
      rec[(size_t)(row0 + kh * 8 + v) * C_DIM + n] = acc[t][v];
  }
}

// ---------------------------------------------------------------------------
extern "C" void kernel_launch(void* const* d_in, const int* in_sizes, int n_in,
                              void* d_out, int out_size, void* d_ws, size_t ws_size,
                              hipStream_t stream) {
  (void)n_in; (void)out_size; (void)ws_size;
  const float* face   = (const float*)d_in[0];
  const float* speech = (const float*)d_in[1];
  const float* svm    = (const float*)d_in[2]; // speech_value_memory [S,C]
  const float* fkm    = (const float*)d_in[3]; // face_key_memory [S,C]
  const int C = C_DIM;
  const int N = in_sizes[0] / C;   // 8192
  const int S = in_sizes[2] / C;   // 4096

  float* out = (float*)d_out;
  float* out_speech_emb = out;
  float* out_speech_rec = out + (size_t)N * C;
  float* out_face_emb   = out + 2 * (size_t)N * C;
  float* out_face_rec   = out + 3 * (size_t)N * C;
  float* out_sa_log     = out + 4 * (size_t)N * C;    // [N,S]
  float* out_fa_log     = out_sa_log + (size_t)N * S; // [N,S]

  // workspace carve-out (~16.5 MB)
  char* ws = (char*)d_ws;
  size_t wo = 0;
  auto carve = [&](size_t bytes) -> char* {
    char* p = ws + wo; wo += (bytes + 255) & ~(size_t)255; return p;
  };
  bf16* speech_b = (bf16*)carve((size_t)N * C * 2);
  bf16* face_b   = (bf16*)carve((size_t)N * C * 2);
  bf16* svm_b    = (bf16*)carve((size_t)S * C * 2);
  bf16* fkm_b    = (bf16*)carve((size_t)S * C * 2);
  bf16* svmT_hi  = (bf16*)carve((size_t)C * S * 2);
  bf16* svmT_lo  = (bf16*)carve((size_t)C * S * 2);
  float* nSp  = (float*)carve((size_t)N * 4);
  float* nFa  = (float*)carve((size_t)N * 4);
  float* nSvm = (float*)carve((size_t)S * 4);
  float* nFkm = (float*)carve((size_t)S * 4);

  // passthrough embeddings
  hipMemcpyAsync(out_speech_emb, speech, (size_t)N * C * 4, hipMemcpyDeviceToDevice, stream);
  hipMemcpyAsync(out_face_emb,   face,   (size_t)N * C * 4, hipMemcpyDeviceToDevice, stream);

  // bf16 casts + row norms
  cast_norm_kernel<<<N, 64, 0, stream>>>(speech, speech_b, nSp, N);
  cast_norm_kernel<<<N, 64, 0, stream>>>(face,   face_b,   nFa, N);
  cast_norm_kernel<<<S, 64, 0, stream>>>(svm,    svm_b,    nSvm, S);
  cast_norm_kernel<<<S, 64, 0, stream>>>(fkm,    fkm_b,    nFkm, S);

  int tc = S * C;
  transpose_split_kernel<<<(tc + 255) / 256, 256, 0, stream>>>(svm, svmT_hi, svmT_lo, S);

  // cosine logits (raw) into the address_log output regions
  int waves1  = (N / 16) * (S / 64);
  int blocks1 = (waves1 + 7) / 8; // 256 threads = 8 waves
  logits_wmma_kernel<<<blocks1, 256, 0, stream>>>(speech_b, svm_b, nSp, nSvm, out_sa_log, N, S);
  logits_wmma_kernel<<<blocks1, 256, 0, stream>>>(face_b,   fkm_b, nFa, nFkm, out_fa_log, N, S);

  // raw logits -> log_softmax, in place
  row_logsoftmax_kernel<<<N, 256, 0, stream>>>(out_sa_log, S);
  row_logsoftmax_kernel<<<N, 256, 0, stream>>>(out_fa_log, S);

  // recall GEMMs (both recalls read speech_value_memory); 1 block per 16-row strip
  int blocks3 = 2 * (N / 16);
  recall_wmma_kernel<<<blocks3, 128, 0, stream>>>(out_sa_log, out_speech_rec,
                                                  out_fa_log, out_face_rec,
                                                  svmT_hi, svmT_lo, N, S);
}
